// MicroLiquidClassifier_32667521253578
// MI455X (gfx1250) — matrix operand order repaired and checked by reading, not compiled, not run on hardware
//
#include <hip/hip_runtime.h>
#include <hip/hip_bf16.h>

// ---------------------------------------------------------------------------
// MicroLiquidClassifier on gfx1250 (MI455X), bf16 WMMA everywhere.
//   B=128, S=1024, D=256, H=512, C=128
// Scan optimizations this round:
//   - W_tau_m strip (16 cols x 512 K) held in 128 VGPRs for all 1024 steps
//   - f32 membrane state lives entirely in registers (8 f32 / lane)
//   - xin/xtau loads hoisted ahead of the WMMA chain + global_prefetch of the
//     next timestep's tile
// ---------------------------------------------------------------------------

typedef __attribute__((ext_vector_type(16))) __bf16 v16bf;
typedef __attribute__((ext_vector_type(8)))  __bf16 v8bf;
typedef __attribute__((ext_vector_type(8)))  float  v8f;

constexpr int kB = 128;
constexpr int kS = 1024;
constexpr int kD = 256;
constexpr int kH = 512;
constexpr int kC = 128;

// ---------------- workspace layout (bytes) ----------------
constexpr size_t OFF_MBF0 = 0;                                   // bf16 mem ping
constexpr size_t SZ_MBF   = (size_t)kB * kH * 2;                 // 131072
constexpr size_t OFF_MBF1 = OFF_MBF0 + SZ_MBF;                   // bf16 mem pong
constexpr size_t OFF_CNT  = OFF_MBF1 + SZ_MBF;                   // u32 spike count
constexpr size_t OFF_BAR  = OFF_CNT + 128;                       // u32 grid barrier
constexpr size_t OFF_ZERO_END = OFF_BAR + 128;                   // zeroed each launch
constexpr size_t OFF_XBF  = OFF_ZERO_END;                        // bf16 x [S][B][D]
constexpr size_t SZ_XBF   = (size_t)kS * kB * kD * 2;            // 64 MiB
constexpr size_t OFF_WIT  = OFF_XBF + SZ_XBF;                    // bf16 W_in^T  [H][D]
constexpr size_t OFF_WTXT = OFF_WIT  + (size_t)kH * kD * 2;      // bf16 Wtau_x^T[H][D]
constexpr size_t OFF_WTMT = OFF_WTXT + (size_t)kH * kD * 2;      // bf16 Wtau_m^T[H][H]
constexpr size_t OFF_WROT = OFF_WTMT + (size_t)kH * kH * 2;      // bf16 W_ro^T  [C][H]
constexpr size_t OFF_XIN  = OFF_WROT + (size_t)kC * kH * 2;      // f32 xin  [S][B][H]
constexpr size_t SZ_ACT   = (size_t)kS * kB * kH * 4;            // 256 MiB
constexpr size_t OFF_XTAU = OFF_XIN + SZ_ACT;                    // f32 xtau [S][B][H]

// ---------------- WMMA fragment loaders ----------------
// A (16x32 bf16, row-major, lda elems): lane m in [0,16) holds row m, K {0..7,16..23};
// lane m+16 holds row m, K {8..15,24..31}.
__device__ __forceinline__ v16bf load_a_frag(const __bf16* tile, int lda, int k0, int lane) {
  const int m = lane & 15, sel = lane >> 4;
  const __bf16* p = tile + (size_t)m * lda + k0 + 8 * sel;
  v8bf lo = *reinterpret_cast<const v8bf*>(p);
  v8bf hi = *reinterpret_cast<const v8bf*>(p + 16);
  v16bf f;
#pragma unroll
  for (int i = 0; i < 8; ++i) { f[i] = lo[i]; f[8 + i] = hi[i]; }
  return f;
}

// B (32x16 bf16) from a transposed weight buffer Wt[n][k] (ldb elems per column):
// lane n in [0,16) holds column n, K {0..7,16..23}; lane n+16 holds column n, K {8..15,24..31}.
__device__ __forceinline__ v16bf load_b_frag(const __bf16* tileT, int ldb, int k0, int lane) {
  const int n = lane & 15, sel = lane >> 4;
  const __bf16* p = tileT + (size_t)n * ldb + k0 + 8 * sel;
  v8bf lo = *reinterpret_cast<const v8bf*>(p);
  v8bf hi = *reinterpret_cast<const v8bf*>(p + 16);
  v16bf f;
#pragma unroll
  for (int i = 0; i < 8; ++i) { f[i] = lo[i]; f[8 + i] = hi[i]; }
  return f;
}

__device__ __forceinline__ v8f wmma_bf16(v16bf a, v16bf b, v8f c) {
  return __builtin_amdgcn_wmma_f32_16x16x32_bf16(false, a, false, b, (short)0, c, false, false);
}

// ---------------- prep kernels ----------------
// x [B,S,D] f32  ->  Xbf [S,B,D] bf16  (writes coalesced)
__global__ void k_convert_x(const float* __restrict__ x, __bf16* __restrict__ Xbf) {
  size_t o = (size_t)blockIdx.x * blockDim.x + threadIdx.x;
  int d = (int)(o % kD);
  int b = (int)((o / kD) % kB);
  int s = (int)(o / ((size_t)kD * kB));
  Xbf[o] = (__bf16)x[(size_t)b * kS * kD + (size_t)s * kD + d];
}

// dst[n*rows_k + k] = (bf16) src[(row0 + k)*src_ld + n]   (transpose + convert)
__global__ void k_transpose_bf16(const float* __restrict__ src, __bf16* __restrict__ dst,
                                 int rows_k, int src_ld, int row0) {
  size_t o = (size_t)blockIdx.x * blockDim.x + threadIdx.x;
  int k = (int)(o % rows_k);
  int n = (int)(o / rows_k);
  dst[o] = (__bf16)src[(size_t)(row0 + k) * src_ld + n];
}

// ---------------- input projection: xin / xtau for all timesteps ----------------
// One wave per (s, mtile, ntile); A fragment shared by both weight GEMMs.
// Store-bound (512 MB of f32 activations), so no deeper tiling needed.
__global__ void k_input_proj(const __bf16* __restrict__ Xbf,
                             const __bf16* __restrict__ WiT,
                             const __bf16* __restrict__ WtxT,
                             const float* __restrict__ b_in,
                             const float* __restrict__ b_tau_lin,
                             const float* __restrict__ b_tau,
                             float* __restrict__ xin, float* __restrict__ xtau) {
  const int wid  = blockIdx.x * (blockDim.x >> 5) + (threadIdx.x >> 5);
  const int lane = threadIdx.x & 31;
  const int nt = wid & 31;          // H/16 = 32 tiles
  const int mt = (wid >> 5) & 7;    // B/16 = 8 tiles
  const int s  = wid >> 8;          // S timesteps

  const __bf16* A  = Xbf + (size_t)s * kB * kD + (size_t)mt * 16 * kD;
  const __bf16* B1 = WiT  + (size_t)nt * 16 * kD;
  const __bf16* B2 = WtxT + (size_t)nt * 16 * kD;

  v8f acc1 = {}, acc2 = {};
#pragma unroll
  for (int k0 = 0; k0 < kD; k0 += 32) {
    v16bf a  = load_a_frag(A, kD, k0, lane);
    v16bf b1 = load_b_frag(B1, kD, k0, lane);
    v16bf b2 = load_b_frag(B2, kD, k0, lane);
    acc1 = wmma_bf16(a, b1, acc1);
    acc2 = wmma_bf16(a, b2, acc2);
  }

  const int n   = nt * 16 + (lane & 15);
  const int sel = lane >> 4;
  const size_t base = (size_t)s * kB * kH;
  const float bi = b_in[n];
  const float bt = b_tau_lin[n] + b_tau[n];
#pragma unroll
  for (int r = 0; r < 8; ++r) {
    const int m = mt * 16 + r + 8 * sel;
    xin [base + (size_t)m * kH + n] = acc1[r] + bi;
    xtau[base + (size_t)m * kH + n] = acc2[r] + bt;
  }
}

// ---------------- grid barrier (persistent scan) ----------------
__device__ __forceinline__ void grid_barrier(unsigned* ctr, unsigned target) {
  __syncthreads();
  if (threadIdx.x == 0) {
    __threadfence();
    __hip_atomic_fetch_add(ctr, 1u, __ATOMIC_RELEASE, __HIP_MEMORY_SCOPE_AGENT);
    while (__hip_atomic_load(ctr, __ATOMIC_ACQUIRE, __HIP_MEMORY_SCOPE_AGENT) < target) {
      __builtin_amdgcn_s_sleep(2);
    }
  }
  __syncthreads();
}

// ---------------- recurrent scan ----------------
// 32 blocks x 256 threads = 256 waves = 256 (16x16) tiles of the [128,512] state.
// Per step: 16 x v_wmma against register-resident W_tau_m fragments, fused
// sigmoid/leak/spike epilogue on register-resident f32 membrane state, bf16
// state double-buffered in global memory, one grid barrier per step.
__global__ void k_scan(const float* __restrict__ xin, const float* __restrict__ xtau,
                       const __bf16* __restrict__ WtmT,
                       __bf16* __restrict__ mbf0, __bf16* __restrict__ mbf1,
                       unsigned* __restrict__ cnt, unsigned* __restrict__ bar) {
  const int wid  = blockIdx.x * (blockDim.x >> 5) + (threadIdx.x >> 5); // 0..255
  const int lane = threadIdx.x & 31;
  const int nt = wid & 31;   // H tile
  const int mt = wid >> 5;   // B tile
  const int n   = nt * 16 + (lane & 15);
  const int sel = lane >> 4;
  const unsigned nblocks = gridDim.x;

  // flat offset of this lane's r=0 element inside the [B][H] state
  const size_t idx0 = (size_t)(mt * 16 + 8 * sel) * kH + n;

  // --- W_tau_m strip for this wave: resident in VGPRs for the whole scan ---
  v16bf bfrag[16];
  {
    const __bf16* Bcol = WtmT + (size_t)nt * 16 * kH;
#pragma unroll
    for (int kk = 0; kk < 16; ++kk) bfrag[kk] = load_b_frag(Bcol, kH, kk * 32, lane);
  }

  // --- f32 membrane state: register-resident for the whole scan ---
  float memv[8];
#pragma unroll
  for (int r = 0; r < 8; ++r) memv[r] = 0.0f;

  unsigned my_spikes = 0;

  for (int s = 0; s < kS; ++s) {
    const __bf16* Abuf = (s & 1) ? mbf1 : mbf0;
    __bf16*       Obuf = (s & 1) ? mbf0 : mbf1;
    const __bf16* A = Abuf + (size_t)mt * 16 * kH;
    const size_t base = (size_t)s * kB * kH;

    // prefetch next timestep's epilogue operands (hits L2/WGP$ next step)
    if (s + 1 < kS) {
      const size_t nbase = base + (size_t)kB * kH;
#pragma unroll
      for (int r = 0; r < 8; ++r) {
        __builtin_prefetch(&xin [nbase + idx0 + (size_t)r * kH], 0, 3);
        __builtin_prefetch(&xtau[nbase + idx0 + (size_t)r * kH], 0, 3);
      }
    }

    // issue this step's epilogue loads early: they overlap the WMMA chain
    float xtv[8], xiv[8];
#pragma unroll
    for (int r = 0; r < 8; ++r) {
      xtv[r] = xtau[base + idx0 + (size_t)r * kH];
      xiv[r] = xin [base + idx0 + (size_t)r * kH];
    }

    v8f acc = {};
#pragma unroll
    for (int kk = 0; kk < 16; ++kk) {
      v16bf a = load_a_frag(A, kH, kk * 32, lane);
      acc = wmma_bf16(a, bfrag[kk], acc);
    }

    unsigned c = 0;
#pragma unroll
    for (int r = 0; r < 8; ++r) {
      float z    = acc[r] + xtv[r];
      float beta = 1.0f / (1.0f + __expf(-z));
      beta = fminf(fmaxf(beta, 0.01f), 0.99f);
      float mn = beta * memv[r] + xiv[r];
      if (mn > 1.0f) { mn -= 1.0f; ++c; }   // spike + reset (THRESHOLD = 1)
      memv[r] = mn;
      Obuf[idx0 + (size_t)r * kH] = (__bf16)mn;
    }
    my_spikes += c;

    grid_barrier(bar, (unsigned)(s + 1) * nblocks);
  }

  // integer spike count -> deterministic across replays
#pragma unroll
  for (int off = 16; off > 0; off >>= 1)
    my_spikes += (unsigned)__shfl_down((int)my_spikes, off, 32);
  if (lane == 0 && my_spikes) atomicAdd(cnt, my_spikes);
}

// ---------------- readout: out = mem @ W_ro + b_ro ; sparsity scalar ----------------
__global__ void k_readout(const __bf16* __restrict__ memf, const __bf16* __restrict__ WroT,
                          const float* __restrict__ b_ro, const unsigned* __restrict__ cnt,
                          float* __restrict__ out) {
  const int wid  = blockIdx.x * (blockDim.x >> 5) + (threadIdx.x >> 5); // 0..63
  const int lane = threadIdx.x & 31;
  const int nt = wid & 7;    // C/16 = 8
  const int mt = wid >> 3;   // B/16 = 8
  const __bf16* A  = memf + (size_t)mt * 16 * kH;
  const __bf16* Bc = WroT + (size_t)nt * 16 * kH;

  v8f acc = {};
#pragma unroll
  for (int k0 = 0; k0 < kH; k0 += 32) {
    v16bf a = load_a_frag(A, kH, k0, lane);
    v16bf b = load_b_frag(Bc, kH, k0, lane);
    acc = wmma_bf16(a, b, acc);
  }

  const int n   = nt * 16 + (lane & 15);
  const int sel = lane >> 4;
#pragma unroll
  for (int r = 0; r < 8; ++r) {
    const int m = mt * 16 + r + 8 * sel;
    out[(size_t)m * kC + n] = acc[r] + b_ro[n];
  }
  if (blockIdx.x == 0 && threadIdx.x == 0) {
    out[(size_t)kB * kC] =
        1.0f - (float)cnt[0] * (1.0f / ((float)kS * (float)kB * (float)kH));
  }
}

// ---------------------------------------------------------------------------
extern "C" void kernel_launch(void* const* d_in, const int* in_sizes, int n_in,
                              void* d_out, int out_size, void* d_ws, size_t ws_size,
                              hipStream_t stream) {
  (void)in_sizes; (void)n_in; (void)out_size; (void)ws_size;

  const float* x         = (const float*)d_in[0];
  const float* W_in      = (const float*)d_in[1];
  const float* b_in      = (const float*)d_in[2];
  const float* W_tau     = (const float*)d_in[3];
  const float* b_tau_lin = (const float*)d_in[4];
  const float* b_tau     = (const float*)d_in[5];
  const float* W_ro      = (const float*)d_in[6];
  const float* b_ro      = (const float*)d_in[7];
  float* out = (float*)d_out;

  char* ws = (char*)d_ws;
  __bf16*   mbf0 = (__bf16*)  (ws + OFF_MBF0);
  __bf16*   mbf1 = (__bf16*)  (ws + OFF_MBF1);
  unsigned* cnt  = (unsigned*)(ws + OFF_CNT);
  unsigned* bar  = (unsigned*)(ws + OFF_BAR);
  __bf16*   Xbf  = (__bf16*)  (ws + OFF_XBF);
  __bf16*   WiT  = (__bf16*)  (ws + OFF_WIT);
  __bf16*   WtxT = (__bf16*)  (ws + OFF_WTXT);
  __bf16*   WtmT = (__bf16*)  (ws + OFF_WTMT);
  __bf16*   WroT = (__bf16*)  (ws + OFF_WROT);
  float*    xin  = (float*)   (ws + OFF_XIN);
  float*    xtau = (float*)   (ws + OFF_XTAU);

  // zero mutable state (bf16 ping-pong, spike count, barrier) — capture-safe
  hipMemsetAsync(ws, 0, OFF_ZERO_END, stream);

  // --- prep: convert / transpose to bf16 ---
  {
    size_t N = (size_t)kS * kB * kD;
    k_convert_x<<<(unsigned)(N / 256), 256, 0, stream>>>(x, Xbf);
  }
  k_transpose_bf16<<<(kD * kH) / 256, 256, 0, stream>>>(W_in,  WiT,  kD, kH, 0);
  k_transpose_bf16<<<(kD * kH) / 256, 256, 0, stream>>>(W_tau, WtxT, kD, kH, 0);
  k_transpose_bf16<<<(kH * kH) / 256, 256, 0, stream>>>(W_tau, WtmT, kH, kH, kD);
  k_transpose_bf16<<<(kH * kC) / 256, 256, 0, stream>>>(W_ro,  WroT, kH, kC, 0);

  // --- big parallel GEMM: input currents + tau logits for all timesteps ---
  {
    int waves  = kS * (kB / 16) * (kH / 16);  // 262144
    int blocks = waves / 8;                   // 8 waves / 256-thread block
    k_input_proj<<<blocks, 256, 0, stream>>>(Xbf, WiT, WtxT, b_in, b_tau_lin, b_tau,
                                             xin, xtau);
  }

  // --- persistent recurrent scan (32 blocks x 8 waves = 256 tiles) ---
  k_scan<<<32, 256, 0, stream>>>(xin, xtau, WtmT, mbf0, mbf1, cnt, bar);

  // --- readout (final bf16 state lives in mbf0 since S is even) ---
  k_readout<<<8, 256, 0, stream>>>(mbf0, WroT, b_ro, cnt, out);
}
